// LSTMModel_83966610637190
// MI455X (gfx1250) — matrix operand order repaired and checked by reading, not compiled, not run on hardware
//
#include <hip/hip_runtime.h>
#include <hip/hip_bf16.h>
#include <math.h>

// Problem dims
#define B_   64
#define T_   256
#define IN_  512
#define H_   1024
#define OUT_ 512
#define KCAT (IN_ + H_)   // 1536 : fused [x | h] K-dim
#define G4   (4 * H_)     // 4096 : fused gate width

#define CHUNK_K 128               // K values staged per chunk (4 WMMA k-steps)
#define NCHUNK  (KCAT / CHUNK_K)  // 12
#define CHUNK_US 8192             // ushorts per chunk buffer: 4 ksteps*4 gates*32 lanes*16

typedef __attribute__((ext_vector_type(16))) __bf16 v16bf;
typedef __attribute__((ext_vector_type(8)))  __bf16 v8bf;
typedef __attribute__((ext_vector_type(8)))  float  v8f;

__device__ __forceinline__ unsigned short f2bf(float f) {
  union { float f; unsigned u; } v; v.f = f;
  unsigned u = v.u;
  u += 0x7FFFu + ((u >> 16) & 1u);   // round-to-nearest-even
  return (unsigned short)(u >> 16);
}

__device__ __forceinline__ float sigmoidf_(float x) {
  return 1.0f / (1.0f + __expf(-x));
}

__device__ __forceinline__ v8f wmma_bf16(v16bf a, v16bf b, v8f c) {
  return __builtin_amdgcn_wmma_f32_16x16x32_bf16(
      false, a, false, b, (short)0, c, false, false);
}

__device__ __forceinline__ v16bf cat16(v8bf lo, v8bf hi) {
  return __builtin_shufflevector(lo, hi, 0,1,2,3,4,5,6,7,8,9,10,11,12,13,14,15);
}

// 16B async copy global -> LDS (ASYNCcnt-tracked, lane-granular)
__device__ __forceinline__ void async_ld16(unsigned ldsoff, const unsigned short* gptr) {
  asm volatile("global_load_async_to_lds_b128 %0, %1, off"
               :: "v"(ldsoff), "v"(gptr) : "memory");
}
__device__ __forceinline__ void wait_async8() {
  asm volatile("s_wait_asynccnt 0x8" ::: "memory");
}
__device__ __forceinline__ void wait_async0() {
  asm volatile("s_wait_asynccnt 0x0" ::: "memory");
}

// ---------------- prep kernels ----------------

__global__ void k_build_wcat(const float* __restrict__ Wi,
                             const float* __restrict__ Wh,
                             unsigned short* __restrict__ wcat) {
  int idx = blockIdx.x * blockDim.x + threadIdx.x;
  const int total = G4 * KCAT;
  for (; idx < total; idx += gridDim.x * blockDim.x) {
    int n = idx / KCAT, k = idx - n * KCAT;
    float v = (k < IN_) ? Wi[(size_t)k * G4 + n]
                        : Wh[(size_t)(k - IN_) * G4 + n];
    wcat[idx] = f2bf(v);
  }
}

__global__ void k_build_woT(const float* __restrict__ Wo,
                            unsigned short* __restrict__ woT) {
  int idx = blockIdx.x * blockDim.x + threadIdx.x;
  const int total = OUT_ * H_;
  for (; idx < total; idx += gridDim.x * blockDim.x) {
    int o = idx / H_, k = idx - o * H_;
    woT[idx] = f2bf(Wo[(size_t)k * OUT_ + o]);
  }
}

__global__ void k_cvt_xs(const float* __restrict__ xs,
                         unsigned short* __restrict__ xsb) {
  int idx = blockIdx.x * blockDim.x + threadIdx.x;
  const int total = B_ * T_ * IN_;
  for (; idx < total; idx += gridDim.x * blockDim.x) xsb[idx] = f2bf(xs[idx]);
}

__global__ void k_zero_state(unsigned short* __restrict__ zs,
                             float* __restrict__ cst) {
  int idx = blockIdx.x * blockDim.x + threadIdx.x;
  if (idx < B_ * H_) { zs[idx] = 0; cst[idx] = 0.0f; }
}

// ---------------- recurrent step ----------------
// Block owns hidden slice j0..j0+15; computes all 4 gates for all 64 batch
// rows, then the fused LSTM elementwise update. B (weight) fragments are
// staged once per block into LDS via async copies (double buffered) and
// shared by all 4 waves; A streams from global (L2-resident).
__global__ __launch_bounds__(128) void k_lstm_step(
    const unsigned short* __restrict__ xsb,   // [B][T][IN] bf16
    const unsigned short* __restrict__ wcat,  // [4096][1536] bf16
    const float* __restrict__ bias,           // [4096]
    unsigned short* __restrict__ zs,          // [T+1][B][H] bf16
    float* __restrict__ cst,                  // [B][H] f32
    int t) {
  __shared__ unsigned short ldsB[2 * CHUNK_US] __attribute__((aligned(32)));
  __shared__ float gl[64][68];

  const int tid  = threadIdx.x;
  const int wave = tid >> 5;                  // 0..3 -> M tile
  const int lane = tid & 31;
  const int lh   = lane >> 4;
  const int l16  = lane & 15;
  const int j0   = blockIdx.x * 16;

  const int brow = wave * 16 + l16;
  const unsigned short* xrow = xsb + (size_t)brow * (T_ * IN_) + (size_t)t * IN_ + lh * 8;
  const unsigned short* hrow = zs + (size_t)t * (B_ * H_) + (size_t)brow * H_ + lh * 8;

  // Producer-side offsets: 1024 16B units per chunk, 8 per thread.
  // unit u -> half h(u&1), lane l((u>>1)&31), gate g((u>>6)&3), kstep s(u>>8)
  unsigned ldsu[8];
  const unsigned short* gbase[8];
#pragma unroll
  for (int i = 0; i < 8; ++i) {
    int u  = tid + i * 128;
    int hh = u & 1;
    int l  = (u >> 1) & 31;
    int g  = (u >> 6) & 3;
    int s  = u >> 8;
    gbase[i] = wcat + (size_t)(g * H_ + j0 + (l & 15)) * KCAT
                    + s * 32 + (l >> 4) * 8 + hh * 16;
    ldsu[i] = (unsigned)(u * 16);
  }
  const unsigned lbase0 = (unsigned)(size_t)(&ldsB[0]);            // LDS byte addr
  const unsigned lbase1 = lbase0 + CHUNK_US * 2;

  v8f a0 = {}, a1 = {}, a2 = {}, a3 = {};

  // prologue: chunk 0 -> buf 0
#pragma unroll
  for (int i = 0; i < 8; ++i) async_ld16(lbase0 + ldsu[i], gbase[i]);

  for (int c = 0; c < NCHUNK; ++c) {
    if (c + 1 < NCHUNK) {
      const unsigned nb = ((c + 1) & 1) ? lbase1 : lbase0;
      const int nkk = (c + 1) * CHUNK_K;
#pragma unroll
      for (int i = 0; i < 8; ++i) async_ld16(nb + ldsu[i], gbase[i] + nkk);
      wait_async8();     // previous chunk's 8 copies retired (in-order)
    } else {
      wait_async0();
    }
    __syncthreads();     // chunk c visible to all waves

    const unsigned short* abase =
        (c < IN_ / CHUNK_K) ? (xrow + c * CHUNK_K) : (hrow + (c * CHUNK_K - IN_));
    const int cb = (c & 1) * CHUNK_US;
#pragma unroll
    for (int s = 0; s < 4; ++s) {
      const unsigned short* ap = abase + s * 32;
      v16bf a = cat16(*(const v8bf*)ap, *(const v8bf*)(ap + 16));
      const int bu = cb + (s * 4 * 32 + lane) * 16;      // ushort index
      v16bf b0 = *(const v16bf*)&ldsB[bu + 0 * 512];
      v16bf b1 = *(const v16bf*)&ldsB[bu + 1 * 512];
      v16bf b2 = *(const v16bf*)&ldsB[bu + 2 * 512];
      v16bf b3 = *(const v16bf*)&ldsB[bu + 3 * 512];
      a0 = wmma_bf16(a, b0, a0);
      a1 = wmma_bf16(a, b1, a1);
      a2 = wmma_bf16(a, b2, a2);
      a3 = wmma_bf16(a, b3, a3);
    }
    __syncthreads();     // all waves done with buf before it is refilled
  }

  // spill gates (+bias) to LDS: cols [0,16)=i [16,32)=f [32,48)=g [48,64)=o
  const float bi = bias[0 * H_ + j0 + l16];
  const float bf = bias[1 * H_ + j0 + l16];
  const float bg = bias[2 * H_ + j0 + l16];
  const float bb = bias[3 * H_ + j0 + l16];
#pragma unroll
  for (int r = 0; r < 8; ++r) {
    int m = wave * 16 + lh * 8 + r;
    gl[m][ 0 + l16] = a0[r] + bi;
    gl[m][16 + l16] = a1[r] + bf;
    gl[m][32 + l16] = a2[r] + bg;
    gl[m][48 + l16] = a3[r] + bb;
  }
  __syncthreads();

  // elementwise LSTM update: 64 rows x 16 hidden / 128 threads
#pragma unroll
  for (int e = 0; e < 8; ++e) {
    int u = e * 128 + tid;
    int row = u >> 4, j = u & 15;
    float vi = sigmoidf_(gl[row][ 0 + j]);
    float vf = sigmoidf_(gl[row][16 + j]);
    float vg = tanhf(gl[row][32 + j]);
    float vo = sigmoidf_(gl[row][48 + j]);
    size_t ci = (size_t)row * H_ + j0 + j;
    float c = vf * cst[ci] + vi * vg;
    cst[ci] = c;
    float h = vo * tanhf(c);
    zs[(size_t)(t + 1) * (B_ * H_) + ci] = f2bf(h);
  }
}

// ---------------- output projection ----------------
__global__ __launch_bounds__(256) void k_proj(
    const unsigned short* __restrict__ zs,   // slab t+1 = h_t
    const unsigned short* __restrict__ woT,  // [OUT][H] bf16
    const float* __restrict__ bo,
    float* __restrict__ out) {
  const int tid  = threadIdx.x;
  const int wave = tid >> 5;
  const int lane = tid & 31;
  const int lh   = lane >> 4;
  const int l16  = lane & 15;
  const int wm   = wave & 3;
  const int wn   = wave >> 2;
  const int t    = blockIdx.y;
  const int colbase = blockIdx.x * 128 + wn * 64;

  const int brow = wm * 16 + l16;
  const unsigned short* arow =
      zs + (size_t)(t + 1) * (B_ * H_) + (size_t)brow * H_ + lh * 16;
  const unsigned short* wb = woT + (size_t)(colbase + l16) * H_ + lh * 16;

  v8f a0 = {}, a1 = {}, a2 = {}, a3 = {};
  for (int kk = 0; kk < H_; kk += 32) {
    v16bf a = cat16(*(const v8bf*)(arow + kk), *(const v8bf*)(arow + kk + 8));
    v16bf b0 = *(const v16bf*)(wb + (size_t) 0 * H_ + kk);
    v16bf b1 = *(const v16bf*)(wb + (size_t)16 * H_ + kk);
    v16bf b2 = *(const v16bf*)(wb + (size_t)32 * H_ + kk);
    v16bf b3 = *(const v16bf*)(wb + (size_t)48 * H_ + kk);
    a0 = wmma_bf16(a, b0, a0);
    a1 = wmma_bf16(a, b1, a1);
    a2 = wmma_bf16(a, b2, a2);
    a3 = wmma_bf16(a, b3, a3);
  }

#pragma unroll
  for (int r = 0; r < 8; ++r) {
    int b = wm * 16 + lh * 8 + r;
    float* orow = out + (size_t)b * (T_ * OUT_) + (size_t)t * OUT_ + colbase;
    orow[ 0 + l16] = a0[r] + bo[colbase +  0 + l16];
    orow[16 + l16] = a1[r] + bo[colbase + 16 + l16];
    orow[32 + l16] = a2[r] + bo[colbase + 32 + l16];
    orow[48 + l16] = a3[r] + bo[colbase + 48 + l16];
  }
}

// ---------------- host launcher ----------------
extern "C" void kernel_launch(void* const* d_in, const int* in_sizes, int n_in,
                              void* d_out, int out_size, void* d_ws, size_t ws_size,
                              hipStream_t stream) {
  const float* xs = (const float*)d_in[0];
  const float* Wi = (const float*)d_in[1];
  const float* Wh = (const float*)d_in[2];
  const float* b  = (const float*)d_in[3];
  const float* Wo = (const float*)d_in[4];
  const float* bo = (const float*)d_in[5];
  float* out = (float*)d_out;

  char* ws = (char*)d_ws;
  size_t off = 0;
  unsigned short* wcat = (unsigned short*)(ws + off); off += (size_t)G4 * KCAT * 2;
  unsigned short* woT  = (unsigned short*)(ws + off); off += (size_t)OUT_ * H_ * 2;
  unsigned short* xsb  = (unsigned short*)(ws + off); off += (size_t)B_ * T_ * IN_ * 2;
  unsigned short* zs   = (unsigned short*)(ws + off); off += (size_t)(T_ + 1) * B_ * H_ * 2;
  float*          cst  = (float*)(ws + off);          off += (size_t)B_ * H_ * 4;

  k_build_wcat<<<1024, 256, 0, stream>>>(Wi, Wh, wcat);
  k_build_woT <<< 512, 256, 0, stream>>>(Wo, woT);
  k_cvt_xs    <<<2048, 256, 0, stream>>>(xs, xsb);
  k_zero_state<<< 256, 256, 0, stream>>>(zs, cst);

  for (int t = 0; t < T_; ++t)
    k_lstm_step<<<H_ / 16, 128, 0, stream>>>(xsb, wcat, b, zs, cst, t);

  k_proj<<<dim3(OUT_ / 128, T_), 256, 0, stream>>>(zs, woT, bo, out);
}
